// Attention_88081189306791
// MI455X (gfx1250) — matrix operand order repaired and checked by reading, not compiled, not run on hardware
//
#include <hip/hip_runtime.h>
#include <hip/hip_bf16.h>
#include <stdint.h>

#define HIDDEN 1024
#define BATCH 32
#define SRC_LEN 2048
#define NEG_BIG -10000000000.0f

typedef __attribute__((ext_vector_type(16))) __bf16 v16bf;
typedef __attribute__((ext_vector_type(8)))  float  v8f;

union FragAB { v16bf v; uint4 q[2]; };

__device__ inline uint32_t pack_bf16x2(float x, float y) {
  uint16_t lx = __builtin_bit_cast(uint16_t, (__bf16)x);
  uint16_t ly = __builtin_bit_cast(uint16_t, (__bf16)y);
  return (uint32_t)lx | ((uint32_t)ly << 16);
}

// Branch-free tanh: native v_tanh_f32 if available, else exp+rcp form that is
// exact at +/-inf (exp(2x)->inf gives 1, ->0 gives -1). Avoids OCML's
// divergent EXEC-masked path seen in round 1.
__device__ inline float fast_tanhf(float x) {
#if __has_builtin(__builtin_amdgcn_tanhf)
  return __builtin_amdgcn_tanhf(x);
#else
  float ex = __expf(2.0f * x);
  return 1.0f - 2.0f * __builtin_amdgcn_rcpf(ex + 1.0f);
#endif
}

// ---------------------------------------------------------------------------
// Kernel 1: pack W_enc (rows 1024..2047 of attn_W, K x N row-major f32) into
// bf16 K-pairs, n-major:  Bpack[n][kp] = { bf16 W[2kp][n], bf16 W[2kp+1][n] }
// LDS-tiled transpose so global reads AND writes are coalesced.
// ---------------------------------------------------------------------------
__global__ __launch_bounds__(256) void pack_Wenc(const float* __restrict__ attn_W,
                                                 uint32_t* __restrict__ Bpack) {
  __shared__ uint16_t tile[64][65];           // [n_local][k_local], padded
  const int t  = threadIdx.x;
  const int k0 = (blockIdx.x >> 4) * 64;      // 16 x 16 tiles of 64x64
  const int n0 = (blockIdx.x & 15) * 64;
  const float* W = attn_W + (size_t)HIDDEN * HIDDEN;   // W_enc base
#pragma unroll
  for (int j = 0; j < 16; ++j) {
    int i = t + 256 * j;
    int r = i >> 6, c = i & 63;               // r = k_local, c = n_local
    float f = W[(size_t)(k0 + r) * HIDDEN + (n0 + c)];
    tile[c][r] = __builtin_bit_cast(uint16_t, (__bf16)f);
  }
  __syncthreads();
#pragma unroll
  for (int j = 0; j < 8; ++j) {
    int i = t + 256 * j;
    int n = i >> 5, dw = i & 31;              // 32 kp per n
    uint32_t lo = tile[n][2 * dw], hi = tile[n][2 * dw + 1];
    Bpack[(size_t)(n0 + n) * (HIDDEN / 2) + (k0 >> 1) + dw] = lo | (hi << 16);
  }
}

// ---------------------------------------------------------------------------
// Kernel 2: dec_proj[b][n] = decoder_hidden[b] . W_dec[:,n] + bias[n]
// ---------------------------------------------------------------------------
__global__ __launch_bounds__(256) void dec_proj_kernel(const float* __restrict__ dec,
                                                       const float* __restrict__ attn_W,
                                                       const float* __restrict__ bias,
                                                       float* __restrict__ dec_proj) {
  int id = blockIdx.x * 256 + threadIdx.x;
  int b = id >> 10, n = id & 1023;
  const float* dh = dec + b * HIDDEN;
  float acc = bias[n];
#pragma unroll 4
  for (int k = 0; k < HIDDEN; ++k)
    acc = fmaf(dh[k], attn_W[(size_t)k * HIDDEN + n], acc);
  dec_proj[id] = acc;
}

// ---------------------------------------------------------------------------
// Kernel 3: fused  logits[b][s] = maskfix( sum_n tanh(enc@W_enc + dec_proj) * v_w )
// WG tile: M=64 s-rows, full N=1024 in two 512 halves, K=1024 in steps of 32.
// 8 waves: (mw 0..1) x (nw 0..3); wave tile 32(M) x 128(N) = 16 WMMA accums.
// ---------------------------------------------------------------------------
__global__ __launch_bounds__(256) void attn_energy_kernel(
    const float* __restrict__ enc,
    const uint32_t* __restrict__ Bpack,
    const float* __restrict__ dec_proj,
    const float* __restrict__ v_w,
    const int* __restrict__ mask,
    float* __restrict__ logits) {
  // padded rows of 20 dwords (16 kp + 4 pad) keep b128 frag loads 16B-aligned
  __shared__ __align__(16) uint32_t Alds[64 * 20];     //  5 KB
  __shared__ __align__(16) uint32_t Blds[512 * 20];    // 40 KB
  __shared__ float attn_lds[64];

  const int t     = threadIdx.x;
  const int lane  = t & 31;
  const int wave  = t >> 5;
  const int mw    = wave >> 2;           // 0..1
  const int nw    = wave & 3;            // 0..3
  const int b     = blockIdx.x >> 5;     // 32 s-blocks per batch
  const int s0    = (blockIdx.x & 31) * 64;
  const int lhalf = (lane < 16) ? 0 : 1;
  const int ln    = lane & 15;

  if (t < 64) attn_lds[t] = 0.0f;
  __syncthreads();

  for (int nh = 0; nh < 2; ++nh) {       // N halves of 512
    v8f acc[2][8];
#pragma unroll
    for (int i = 0; i < 2; ++i)
#pragma unroll
      for (int j = 0; j < 8; ++j) acc[i][j] = {};

    for (int k0 = 0; k0 < HIDDEN; k0 += 32) {
      // --- stage global -> regs (coalesced 128B segments) ---
      float4 af[2];
      uint4  bg[8];
#pragma unroll
      for (int j = 0; j < 2; ++j) {
        int i = t + 256 * j;
        int m = i >> 3, k4 = (i & 7) * 4;
        af[j] = *reinterpret_cast<const float4*>(
            &enc[((size_t)(b * SRC_LEN + s0 + m)) * HIDDEN + k0 + k4]);
      }
#pragma unroll
      for (int j = 0; j < 8; ++j) {
        int i = t + 256 * j;
        int n = i >> 2, q = i & 3;
        bg[j] = *reinterpret_cast<const uint4*>(
            &Bpack[(size_t)(nh * 512 + n) * (HIDDEN / 2) + (k0 >> 1) + q * 4]);
      }
      __syncthreads();                   // previous iter's frag reads done
      // --- regs -> LDS ---
#pragma unroll
      for (int j = 0; j < 2; ++j) {
        int i = t + 256 * j;
        int m = i >> 3, kp = (i & 7) * 2;   // dword index of first pair
        Alds[m * 20 + kp]     = pack_bf16x2(af[j].x, af[j].y);
        Alds[m * 20 + kp + 1] = pack_bf16x2(af[j].z, af[j].w);
      }
#pragma unroll
      for (int j = 0; j < 8; ++j) {
        int i = t + 256 * j;
        int n = i >> 2, q = i & 3;
        *reinterpret_cast<uint4*>(&Blds[n * 20 + q * 4]) = bg[j];
      }
      __syncthreads();

      // --- A fragments (ISA 16-bit A 16x32: lanes<16 kp{0-3,8-11}, >=16 kp{4-7,12-15}) ---
      FragAB a[2];
#pragma unroll
      for (int mt = 0; mt < 2; ++mt) {
        const uint4* Arow =
            reinterpret_cast<const uint4*>(&Alds[(mw * 32 + mt * 16 + ln) * 20]);
        a[mt].q[0] = Arow[lhalf];
        a[mt].q[1] = Arow[lhalf + 2];
      }
      // --- B fragments (lanes<16: kp{0-3,8-11}, lanes>=16: kp{4-7,12-15}; N = lane&15) ---
#pragma unroll
      for (int nt = 0; nt < 8; ++nt) {
        FragAB bf;
        const uint4* Brow =
            reinterpret_cast<const uint4*>(&Blds[(nw * 128 + nt * 16 + ln) * 20]);
        bf.q[0] = Brow[lhalf];
        bf.q[1] = Brow[lhalf + 2];
#pragma unroll
        for (int mt = 0; mt < 2; ++mt) {
          acc[mt][nt] = __builtin_amdgcn_wmma_f32_16x16x32_bf16(
              false, a[mt].v, false, bf.v, (short)0, acc[mt][nt], false, false);
        }
      }
    }

    // --- epilogue: tanh(acc + dec_proj) * v_w, reduce over N into attn_lds ---
    // C layout: lanes<16 M=r, lanes>=16 M=r+8; N = lane&15.
    // Butterfly transpose-reduce across the 16-lane group: 8 shfl per tile
    // instead of 32; lane g ends holding the full sum of row rev3(g&7).
#pragma unroll
    for (int nt = 0; nt < 8; ++nt) {
      int ng   = nh * 512 + nw * 128 + nt * 16 + ln;
      float dp = dec_proj[b * HIDDEN + ng];
      float vw = v_w[ng];
#pragma unroll
      for (int mt = 0; mt < 2; ++mt) {
        float v[8];
#pragma unroll
        for (int r = 0; r < 8; ++r)
          v[r] = fast_tanhf(acc[mt][nt][r] + dp) * vw;
        // step 1 (xor 1): keep rows 0-3 (even) / 4-7 (odd)
#pragma unroll
        for (int j = 0; j < 4; ++j) {
          float send = (ln & 1) ? v[j] : v[j + 4];
          float recv = __shfl_xor(send, 1, 16);
          v[j] = ((ln & 1) ? v[j + 4] : v[j]) + recv;
        }
        // step 2 (xor 2)
#pragma unroll
        for (int j = 0; j < 2; ++j) {
          float send = (ln & 2) ? v[j] : v[j + 2];
          float recv = __shfl_xor(send, 2, 16);
          v[j] = ((ln & 2) ? v[j + 2] : v[j]) + recv;
        }
        // step 3 (xor 4)
        {
          float send = (ln & 4) ? v[0] : v[1];
          float recv = __shfl_xor(send, 4, 16);
          v[0] = ((ln & 4) ? v[1] : v[0]) + recv;
        }
        // step 4 (xor 8): finish the 16-lane sum
        v[0] += __shfl_xor(v[0], 8, 16);
        if ((ln & 8) == 0) {
          int r = ((ln & 1) << 2) | (ln & 2) | ((ln & 4) >> 2);  // bit-reverse
          int m = mw * 32 + mt * 16 + r + lhalf * 8;
          atomicAdd(&attn_lds[m], v[0]);
        }
      }
    }
  }

  __syncthreads();
  if (t < 64) {
    int s = s0 + t;
    float logit = attn_lds[t];
    if (mask[b * SRC_LEN + s] == 0) logit = NEG_BIG;
    logits[b * SRC_LEN + s] = logit;
  }
}

// ---------------------------------------------------------------------------
// Kernel 4: softmax over s per batch row; also zero the context output region.
// ---------------------------------------------------------------------------
__global__ __launch_bounds__(256) void softmax_ctxzero_kernel(
    const float* __restrict__ logits, float* __restrict__ out) {
  __shared__ float red[8];
  const int b = blockIdx.x, t = threadIdx.x;
  const int lane = t & 31, wid = t >> 5;
  float v[8];
  float lm = -3.4e38f;
#pragma unroll
  for (int i = 0; i < 8; ++i) {
    v[i] = logits[b * SRC_LEN + t + i * 256];
    lm = fmaxf(lm, v[i]);
  }
#pragma unroll
  for (int m = 16; m >= 1; m >>= 1) lm = fmaxf(lm, __shfl_xor(lm, m, 32));
  if (lane == 0) red[wid] = lm;
  __syncthreads();
  float M = red[0];
#pragma unroll
  for (int w = 1; w < 8; ++w) M = fmaxf(M, red[w]);
  __syncthreads();
  float ls = 0.f;
#pragma unroll
  for (int i = 0; i < 8; ++i) { v[i] = __expf(v[i] - M); ls += v[i]; }
#pragma unroll
  for (int m = 16; m >= 1; m >>= 1) ls += __shfl_xor(ls, m, 32);
  if (lane == 0) red[wid] = ls;
  __syncthreads();
  float S = 0.f;
#pragma unroll
  for (int w = 0; w < 8; ++w) S += red[w];
  float inv = 1.0f / S;
#pragma unroll
  for (int i = 0; i < 8; ++i) out[b * SRC_LEN + t + i * 256] = v[i] * inv;
  // zero context region (out + BATCH*SRC_LEN)
  float* ctx = out + BATCH * SRC_LEN;
#pragma unroll
  for (int i = 0; i < 4; ++i) ctx[b * HIDDEN + t + i * 256] = 0.f;
}

// ---------------------------------------------------------------------------
// Kernel 5: context[b][h] = sum_s w[b][s] * enc[b][s][h]  (partial + atomic)
// grid: b(32) x hchunk(4) x schunk(8)
// ---------------------------------------------------------------------------
__global__ __launch_bounds__(256) void context_kernel(
    const float* __restrict__ weights, const float* __restrict__ enc,
    float* __restrict__ ctx) {
  const int t   = threadIdx.x;
  const int bid = blockIdx.x;
  const int b   = bid >> 5;
  const int hc  = (bid >> 3) & 3;
  const int sc  = bid & 7;
  const int h   = hc * 256 + t;
  const float* w = weights + b * SRC_LEN + sc * 256;
  const float* e = enc + ((size_t)b * SRC_LEN + sc * 256) * HIDDEN + h;
  float acc = 0.f;
#pragma unroll 4
  for (int s = 0; s < 256; ++s) acc = fmaf(w[s], e[(size_t)s * HIDDEN], acc);
  atomicAdd(&ctx[b * HIDDEN + h], acc);
}

// ---------------------------------------------------------------------------
extern "C" void kernel_launch(void* const* d_in, const int* in_sizes, int n_in,
                              void* d_out, int out_size, void* d_ws, size_t ws_size,
                              hipStream_t stream) {
  const float* dec   = (const float*)d_in[0];
  const float* enc   = (const float*)d_in[1];
  const int*   mask  = (const int*)d_in[2];
  const float* attnW = (const float*)d_in[3];
  const float* attnb = (const float*)d_in[4];
  const float* vw    = (const float*)d_in[5];
  float* out = (float*)d_out;

  uint8_t* ws = (uint8_t*)d_ws;
  uint32_t* Bpack   = (uint32_t*)ws;                                    // 2 MB
  float*    decproj = (float*)(ws + (size_t)HIDDEN * (HIDDEN / 2) * 4); // 128 KB
  float*    logits  = decproj + BATCH * HIDDEN;                         // 256 KB

  pack_Wenc<<<256, 256, 0, stream>>>(attnW, Bpack);
  dec_proj_kernel<<<(BATCH * HIDDEN) / 256, 256, 0, stream>>>(dec, attnW, attnb, decproj);
  attn_energy_kernel<<<BATCH * (SRC_LEN / 64), 256, 0, stream>>>(enc, Bpack, decproj,
                                                                 vw, mask, logits);
  softmax_ctxzero_kernel<<<BATCH, 256, 0, stream>>>(logits, out);
  context_kernel<<<BATCH * 32, 256, 0, stream>>>(out, enc, out + BATCH * SRC_LEN);
}
